// GNNLayer_41300405518568
// MI455X (gfx1250) — compile-verified, matrix operand order checked by reading, not run on hardware
//
#include <hip/hip_runtime.h>

typedef __attribute__((ext_vector_type(16))) _Float16 v16h;
typedef __attribute__((ext_vector_type(8)))  _Float16 v8h;
typedef __attribute__((ext_vector_type(8)))  float    v8f;

#define NNODES   100000
#define KNB      16
#define CIN      128
#define TM       64          // nodes per workgroup (4 M-tiles of 16)
#define NTHREADS 512         // 16 wave32 -> 4 waves/SIMD on one WGP
#define NBLOCKS  ((NNODES + TM - 1) / TM)   // 1563 (last block partial)
#define LSTRIDE  136         // padded f16 row stride (272B = 17*16B -> aligned, bank-conflict-free)

// dynamic LDS partition (in halves):
//   wlds : 1024 rows * 136  (rows 0..511 = W_ih, 512..1023 = W_hh) = 139264 halves
//   xbuf : 64 * 136 = 8704
//   hbuf : 64 * 136 = 8704
#define WLDS_HALVES (1024 * LSTRIDE)
#define BUF_HALVES  (TM * LSTRIDE)
#define SMEM_BYTES  ((WLDS_HALVES + 2 * BUF_HALVES) * 2)   // 313344 B <= 320KB/WG

// 16x32 f16 A/B fragment (ISA 7.12.2 layout) from a row-major source.
// Per-lane pointer: base + (row0 + lane%16)*stride + kc*32 + (lane/16)*8
// halves[0..7] = K+0..7 (16B), halves[8..15] = K+16..23 (16B)
__device__ __forceinline__ v16h frag_from(const _Float16* p) {
    union { v16h v; v8h h[2]; } u;
    u.h[0] = *(const v8h*)(p);
    u.h[1] = *(const v8h*)(p + 16);
    return u.v;
}

__device__ __forceinline__ v8f wmma16(v16h a, v16h b, v8f c) {
    return __builtin_amdgcn_wmma_f32_16x16x32_f16(false, a, false, b, (short)0, c, false, false);
}

// 1 trans exp + 1 trans rcp each; correct saturation at +/-inf, NaN propagates
__device__ __forceinline__ float sigm_fast(float x) {
    return __builtin_amdgcn_rcpf(1.0f + __expf(-x));
}
__device__ __forceinline__ float tanh_fast(float x) {
    return 1.0f - 2.0f * __builtin_amdgcn_rcpf(__expf(2.0f * x) + 1.0f);
}

__global__ void convert_weights_f16(const float* __restrict__ wih,
                                    const float* __restrict__ whh,
                                    const float* __restrict__ ll,
                                    const float* __restrict__ lr,
                                    _Float16* __restrict__ out) {
    int i = blockIdx.x * 256 + threadIdx.x;
    const int S1 = 4 * CIN * CIN;        // 65536
    const int S2 = 2 * S1;               // 131072
    const int S3 = S2 + CIN * CIN;       // 147456
    const int S4 = S3 + CIN * CIN;       // 163840
    if (i < S1)       out[i] = (_Float16)wih[i];
    else if (i < S2)  out[i] = (_Float16)whh[i - S1];
    else if (i < S3)  out[i] = (_Float16)ll[i - S2];
    else if (i < S4)  out[i] = (_Float16)lr[i - S3];
}

__global__ __launch_bounds__(NTHREADS)
void gnn_lstm_sage_kernel(const float*    __restrict__ node_feats,
                          const int*      __restrict__ src_idx,   // edge_index[0], length N*K
                          const _Float16* __restrict__ wih,       // [512,128] f16 row-major
                          const _Float16* __restrict__ whh,       // [512,128]
                          const float*    __restrict__ b_ih,      // [512]
                          const float*    __restrict__ b_hh,      // [512]
                          const _Float16* __restrict__ linl,      // [128,128]
                          const float*    __restrict__ linl_b,    // [128]
                          const _Float16* __restrict__ linr,      // [128,128]
                          float*          __restrict__ out)       // [N,128]
{
    extern __shared__ _Float16 smem[];
    _Float16* wlds = smem;                       // [1024][LSTRIDE]
    _Float16* xbuf = smem + WLDS_HALVES;         // [TM][LSTRIDE]
    _Float16* hbuf = xbuf + BUF_HALVES;          // [TM][LSTRIDE]

    const int tid   = threadIdx.x;
    const int lane  = tid & 31;
    const int wave  = tid >> 5;          // 0..15
    const int lr16  = lane & 15;
    const int hi    = lane >> 4;         // 0/1
    const int mtile = wave >> 2;         // 0..3 : which 16-node tile
    const int qw    = wave & 3;          // 0..3 : channel-group pair owner
    const int nodeBase = blockIdx.x * TM;

    // staging assignment: each thread loads 16 floats of the 64x128 tile
    const int srow = tid >> 3;           // 0..63
    const int scol = (tid & 7) * 16;     // 0..112
    const int myNode  = nodeBase + srow;
    const int clNode  = (myNode < NNODES) ? myNode : (NNODES - 1);  // clamp for partial last block
    const int eBase   = clNode * KNB;

    // ---- stage W_ih / W_hh (f16) into padded LDS rows; zero x/h buffers ----
    for (int r = tid; r < 1024; r += NTHREADS) {
        const _Float16* sp = (r < 512) ? (wih + r * CIN) : (whh + (r - 512) * CIN);
        _Float16* dp = wlds + r * LSTRIDE;
#pragma unroll
        for (int j = 0; j < 16; ++j)                 // 256B per row as 16B chunks
            ((int4*)dp)[j] = ((const int4*)sp)[j];
    }
    for (int i = tid; i < 2 * BUF_HALVES; i += NTHREADS) xbuf[i] = (_Float16)0.0f;

    // per-wave biases for the two owned channel groups (col = lane%16 within group)
    float bi[2], bf[2], bg[2], bo[2], blb[2];
#pragma unroll
    for (int q = 0; q < 2; ++q) {
        const int ch = (qw * 2 + q) * 16 + lr16;
        bi[q]  = b_ih[0 * CIN + ch] + b_hh[0 * CIN + ch];
        bf[q]  = b_ih[1 * CIN + ch] + b_hh[1 * CIN + ch];
        bg[q]  = b_ih[2 * CIN + ch] + b_hh[2 * CIN + ch];
        bo[q]  = b_ih[3 * CIN + ch] + b_hh[3 * CIN + ch];
        blb[q] = linl_b[ch];
    }

    v8f cst0 = {0.f,0.f,0.f,0.f,0.f,0.f,0.f,0.f};  // cell state, group 0
    v8f cst1 = cst0;                               // cell state, group 1

    const int arow = (mtile * 16 + lr16) * LSTRIDE + hi * 8;  // A-frag per-lane base (+kc*32)

    // ---- software-pipelined gather: x_t raw rows live in registers one step ahead ----
    float4 xr[4];
    {
        const int s0 = src_idx[eBase];             // t = 0
        const float* rp = node_feats + (size_t)s0 * CIN + scol;
#pragma unroll
        for (int j = 0; j < 4; ++j) xr[j] = ((const float4*)rp)[j];
    }
    int sNxt = src_idx[eBase + 1];                 // t = 1 index

    __syncthreads();   // weights staged, buffers zeroed

    for (int t = 0; t < KNB; ++t) {
        // ---- commit prefetched x_t into LDS (f32 -> f16) ----
        {
            _Float16* dp = &xbuf[srow * LSTRIDE + scol];
#pragma unroll
            for (int j = 0; j < 4; ++j) {
                dp[4 * j + 0] = (_Float16)xr[j].x;
                dp[4 * j + 1] = (_Float16)xr[j].y;
                dp[4 * j + 2] = (_Float16)xr[j].z;
                dp[4 * j + 3] = (_Float16)xr[j].w;
            }
        }

        // read h_{t-1} A-fragments (hbuf stable since last barrier)
        v16h ah[4];
#pragma unroll
        for (int kc = 0; kc < 4; ++kc)
            ah[kc] = frag_from(&hbuf[arow + kc * 32]);

        __syncthreads();   // x_t staged; all ah reads done before h is overwritten

        v16h ax[4];
#pragma unroll
        for (int kc = 0; kc < 4; ++kc)
            ax[kc] = frag_from(&xbuf[arow + kc * 32]);

        // ---- issue next gather now; latency hides under the WMMA section ----
        if (t + 1 < KNB) {
            const float* rp = node_feats + (size_t)sNxt * CIN + scol;
#pragma unroll
            for (int j = 0; j < 4; ++j) xr[j] = ((const float4*)rp)[j];
            if (t + 2 < KNB) sNxt = src_idx[eBase + t + 2];
        }

#pragma unroll
        for (int q = 0; q < 2; ++q) {
            const int cg = qw * 2 + q;                            // channel group 0..7
            const int wr = (cg * 16 + lr16) * LSTRIDE + hi * 8;   // per-lane LDS weight base
            v8f ai = {0.f,0.f,0.f,0.f,0.f,0.f,0.f,0.f};
            v8f af = ai, ag = ai, ao = ai;
#pragma unroll
            for (int kc = 0; kc < 4; ++kc) {
                const int ko  = kc * 32;
                const int ihb = wr + ko;                          // W_ih rows 0..511
                const int hhb = 512 * LSTRIDE + wr + ko;          // W_hh rows 512..1023
                ai = wmma16(ax[kc], frag_from(wlds + 0 * CIN * LSTRIDE + ihb), ai);
                af = wmma16(ax[kc], frag_from(wlds + 1 * CIN * LSTRIDE + ihb), af);
                ag = wmma16(ax[kc], frag_from(wlds + 2 * CIN * LSTRIDE + ihb), ag);
                ao = wmma16(ax[kc], frag_from(wlds + 3 * CIN * LSTRIDE + ihb), ao);
                ai = wmma16(ah[kc], frag_from(wlds + 0 * CIN * LSTRIDE + hhb), ai);
                af = wmma16(ah[kc], frag_from(wlds + 1 * CIN * LSTRIDE + hhb), af);
                ag = wmma16(ah[kc], frag_from(wlds + 2 * CIN * LSTRIDE + hhb), ag);
                ao = wmma16(ah[kc], frag_from(wlds + 3 * CIN * LSTRIDE + hhb), ao);
            }
            // ---- LSTM elementwise update for this wave's 16 channels x 16 nodes ----
            const int ch = cg * 16 + lr16;
            v8f c = q ? cst1 : cst0;
#pragma unroll
            for (int v = 0; v < 8; ++v) {
                const float iv = sigm_fast(ai[v] + bi[q]);
                const float fv = sigm_fast(af[v] + bf[q]);
                const float gv = tanh_fast(ag[v] + bg[q]);
                const float ov = sigm_fast(ao[v] + bo[q]);
                const float cv = fv * c[v] + iv * gv;
                c[v] = cv;
                const float hv = ov * tanh_fast(cv);
                hbuf[(mtile * 16 + v + 8 * hi) * LSTRIDE + ch] = (_Float16)hv;
            }
            if (q) cst1 = c; else cst0 = c;
        }
        __syncthreads();   // h_t complete; x reads done -> next iter may restage
    }

    // ---- SAGE head: out = relu(h @ linl^T + b + x_self @ linr^T) ----
    {
        const float* rp = node_feats + (size_t)clNode * CIN + scol;
        _Float16* dp = &xbuf[srow * LSTRIDE + scol];
#pragma unroll
        for (int j = 0; j < 4; ++j) {
            float4 f = ((const float4*)rp)[j];
            dp[4 * j + 0] = (_Float16)f.x;
            dp[4 * j + 1] = (_Float16)f.y;
            dp[4 * j + 2] = (_Float16)f.z;
            dp[4 * j + 3] = (_Float16)f.w;
        }
    }
    v16h hfin[4];
#pragma unroll
    for (int kc = 0; kc < 4; ++kc)
        hfin[kc] = frag_from(&hbuf[arow + kc * 32]);
    __syncthreads();
    v16h xfin[4];
#pragma unroll
    for (int kc = 0; kc < 4; ++kc)
        xfin[kc] = frag_from(&xbuf[arow + kc * 32]);

#pragma unroll
    for (int q = 0; q < 2; ++q) {
        const int cg = qw * 2 + q;
        const int wr = (cg * 16 + lr16) * CIN + hi * 8;  // global (unpadded) tail weights
        v8f acc = {0.f,0.f,0.f,0.f,0.f,0.f,0.f,0.f};
#pragma unroll
        for (int kc = 0; kc < 4; ++kc) {
            const int ko = kc * 32;
            acc = wmma16(hfin[kc], frag_from(linl + wr + ko), acc);
            acc = wmma16(xfin[kc], frag_from(linr + wr + ko), acc);
        }
        const int ch = cg * 16 + lr16;
#pragma unroll
        for (int v = 0; v < 8; ++v) {
            const int row = nodeBase + mtile * 16 + v + 8 * hi;
            if (row < NNODES) {
                float o = acc[v] + blb[q];
                o = fmaxf(o, 0.0f);
                out[(size_t)row * CIN + ch] = o;
            }
        }
    }
}

extern "C" void kernel_launch(void* const* d_in, const int* in_sizes, int n_in,
                              void* d_out, int out_size, void* d_ws, size_t ws_size,
                              hipStream_t stream) {
    const float* node_feats = (const float*)d_in[0];
    const int*   edge_index = (const int*)d_in[1];   // [2, E]; row 0 = src
    const float* W_ih       = (const float*)d_in[2];
    const float* W_hh       = (const float*)d_in[3];
    const float* b_ih       = (const float*)d_in[4];
    const float* b_hh       = (const float*)d_in[5];
    const float* lin_l_w    = (const float*)d_in[6];
    const float* lin_l_b    = (const float*)d_in[7];
    const float* lin_r_w    = (const float*)d_in[8];
    float* out = (float*)d_out;

    (void)in_sizes; (void)n_in; (void)out_size; (void)ws_size;

    _Float16* ws     = (_Float16*)d_ws;
    _Float16* wih_h  = ws;                         // 512*128
    _Float16* whh_h  = ws + 4 * CIN * CIN;         // 512*128
    _Float16* linl_h = ws + 8 * CIN * CIN;         // 128*128
    _Float16* linr_h = linl_h + CIN * CIN;         // 128*128

    // allow the large dynamic-LDS allocation (CDNA5: up to 320KB per workgroup)
    (void)hipFuncSetAttribute((const void*)gnn_lstm_sage_kernel,
                              hipFuncAttributeMaxDynamicSharedMemorySize, SMEM_BYTES);

    const int convN = (4 * CIN * CIN) * 2 + 2 * CIN * CIN;   // 163840 elements
    convert_weights_f16<<<(convN + 255) / 256, 256, 0, stream>>>(
        W_ih, W_hh, lin_l_w, lin_r_w, ws);

    gnn_lstm_sage_kernel<<<NBLOCKS, NTHREADS, SMEM_BYTES, stream>>>(
        node_feats, edge_index, wih_h, whh_h, b_ih, b_hh,
        linl_h, lin_l_b, linr_h, out);
}
